// MultiTimeframeLNN_23545010716812
// MI455X (gfx1250) — matrix operand order repaired and checked by reading, not compiled, not run on hardware
//
#include <hip/hip_runtime.h>
#include <hip/hip_bf16.h>
#include <stdint.h>

// ---------------------------------------------------------------------------
// Types for CDNA5 WMMA (gfx1250): bf16 16x16x32, f32 accumulate.
// ---------------------------------------------------------------------------
typedef __attribute__((ext_vector_type(16))) __bf16 v16bf;
typedef __attribute__((ext_vector_type(8)))  float  v8f;

#define DEVINL static __device__ __forceinline__

// Native convert: backend selects the hardware f32->bf16 instruction (RNE).
DEVINL unsigned short f2bf_bits(float f) {
  union { __bf16 h; unsigned short u; } c;
  c.h = (__bf16)f;
  return c.u;
}

DEVINL float sigm(float x) { return 1.0f / (1.0f + __expf(-x)); }

union FragU { v16bf v; uint4 q[2]; };

// A fragment (16xK row-major bf16 in LDS), tile at k0.
// ISA layout: lane<16 holds M=lane, K = k0+{0..7, 16..23}; lane>=16: K = k0+{8..15, 24..31}
DEVINL v16bf loadA(const unsigned short* Ab, int Ks, int k0, int lane) {
  int m = lane & 15;
  int koff = (lane >> 4) << 3;
  const unsigned short* p = Ab + m * Ks + k0 + koff;
  FragU f;
  f.q[0] = *(const uint4*)(p);
  f.q[1] = *(const uint4*)(p + 16);
  return f.v;
}

// B fragment: two 16-byte chunks at p and p+16 (elements) from a precomputed per-lane pointer.
DEVINL v16bf loadBp(const unsigned short* p) {
  FragU f;
  f.q[0] = *(const uint4*)(p);
  f.q[1] = *(const uint4*)(p + 16);
  return f.v;
}

// 16x512 GEMM slab per wave: 4 tiles at columns {w*32, w*32+16, 256+w*32, 256+w*32+16}.
// Software-pipelined: A + 4 B fragments for k0+32 are issued before the k0 WMMAs.
DEVINL void gemm4(const unsigned short* Ab, int Ks,
                  const unsigned short* Wt, int K,
                  int w, int lane, v8f acc[4]) {
  int nbase = (w << 5) + (lane & 15);
  const unsigned short* bp[4];
#pragma unroll
  for (int t = 0; t < 4; ++t) {
    int n = ((t >> 1) << 8) + nbase + ((t & 1) << 4);
    bp[t] = Wt + (size_t)n * K + ((lane >> 4) << 3);
  }
  v16bf a = loadA(Ab, Ks, 0, lane);
  v16bf b[4];
#pragma unroll
  for (int t = 0; t < 4; ++t) b[t] = loadBp(bp[t]);
  for (int k0 = 0; k0 < K; k0 += 32) {
    v16bf an = a;
    v16bf bn[4];
#pragma unroll
    for (int t = 0; t < 4; ++t) bn[t] = b[t];
    if (k0 + 32 < K) {            // prefetch next iteration (uniform branch)
      an = loadA(Ab, Ks, k0 + 32, lane);
#pragma unroll
      for (int t = 0; t < 4; ++t) bn[t] = loadBp(bp[t] + k0 + 32);
    }
#pragma unroll
    for (int t = 0; t < 4; ++t)
      acc[t] = __builtin_amdgcn_wmma_f32_16x16x32_bf16(
          false, a, false, b[t], (short)0, acc[t], false, false);
    a = an;
#pragma unroll
    for (int t = 0; t < 4; ++t) b[t] = bn[t];
  }
}

DEVINL void zero4(v8f acc[4]) {
  v8f z = {0.f, 0.f, 0.f, 0.f, 0.f, 0.f, 0.f, 0.f};
#pragma unroll
  for (int t = 0; t < 4; ++t) acc[t] = z;
}

// Per-step fixed terms: [u@Win | u@Wg_u + bg]  -> fixin/fixg registers.
DEVINL void fixed_gemm(const unsigned short* Ab, int Ks,
                       const unsigned short* W, int K, const float bgv[2],
                       int w, int lane, float fixin[2][8], float fixg[2][8]) {
  v8f acc[4]; zero4(acc);
  gemm4(Ab, Ks, W, K, w, lane, acc);
#pragma unroll
  for (int t = 0; t < 2; ++t) {
#pragma unroll
    for (int r = 0; r < 8; ++r) {
      fixin[t][r] = acc[t][r];
      fixg[t][r]  = acc[2 + t][r] + bgv[t];
    }
  }
}

// Write this lane's 16 state values as bf16 into an A-matrix buffer.
DEVINL void write_hh(unsigned short* buf, const float v[2][8],
                     int n0, int n1, int mbase) {
#pragma unroll
  for (int r = 0; r < 8; ++r) {
    buf[(mbase + r) * 256 + n0] = f2bf_bits(v[0][r]);
    buf[(mbase + r) * 256 + n1] = f2bf_bits(v[1][r]);
  }
}

// RK4 for one LTC cell with double-buffered A matrices: ONE barrier/substep.
// Pre-conditions (established by caller): hh0 (== hb) already written to bufA,
// barrier passed. Buffer alternation guarantees a buffer written at use u is
// re-written only at use u+2, after the barrier of use u+1 (race-free).
DEVINL void rk4_cell(unsigned short* bufA, unsigned short* bufB, float* hstate,
                     const unsigned short* recW, const float tinv[2],
                     const float fixin[2][8], const float fixg[2][8],
                     const float hb[2][8], int w, int lane) {
  int n0 = (w << 5) + (lane & 15), n1 = n0 + 16;
  int mbase = (lane >> 4) << 3;
  float hh[2][8], ac[2][8], kk[2][8];
#pragma unroll
  for (int t = 0; t < 2; ++t)
#pragma unroll
    for (int r = 0; r < 8; ++r) { hh[t][r] = hb[t][r]; ac[t][r] = 0.f; }
  const float aco[4] = {0.f, 0.5f, 0.5f, 1.f};
  const float wco[4] = {1.f, 2.f, 2.f, 1.f};
  const unsigned short* rd = bufA;
  unsigned short* wr = bufB;
#pragma unroll
  for (int i = 0; i < 4; ++i) {
    v8f acc[4]; zero4(acc);
    gemm4(rd, 256, recW, 256, w, lane, acc);
#pragma unroll
    for (int t = 0; t < 2; ++t) {
#pragma unroll
      for (int r = 0; r < 8; ++r) {
        float gpre = acc[t][r] + fixg[t][r];
        float rec  = acc[2 + t][r];
        float k = -hh[t][r] * tinv[t] + fixin[t][r] + sigm(gpre) * rec;
        kk[t][r] = k;
        ac[t][r] += wco[i] * k;
      }
    }
    if (i < 3) {
      float a = aco[i + 1];
#pragma unroll
      for (int t = 0; t < 2; ++t)
#pragma unroll
        for (int r = 0; r < 8; ++r) hh[t][r] = hb[t][r] + a * kk[t][r];
      write_hh(wr, hh, n0, n1, mbase);  // writes buffer NOT being read
    } else {
#pragma unroll
      for (int t = 0; t < 2; ++t) {
        int n = t ? n1 : n0;
#pragma unroll
        for (int r = 0; r < 8; ++r)
          hstate[(mbase + r) * 256 + n] = tanhf(hb[t][r] + ac[t][r] * (1.f / 6.f));
      }
    }
    __syncthreads();
    const unsigned short* tmp = rd; rd = wr; wr = (unsigned short*)tmp;
  }
}

// LayerNorm over 256 cols; wave w handles rows 2w, 2w+1 of a 16x256 f32 LDS buf.
DEVINL void layernorm16(float* buf, const float g[8], const float b[8], int w, int lane) {
#pragma unroll
  for (int rr = 0; rr < 2; ++rr) {
    int m = 2 * w + rr;
    float xv[8];
    float s = 0.f;
#pragma unroll
    for (int j = 0; j < 8; ++j) { xv[j] = buf[m * 256 + lane + 32 * j]; s += xv[j]; }
    for (int o = 16; o; o >>= 1) s += __shfl_xor(s, o, 32);
    float mean = s * (1.f / 256.f);
    float v2 = 0.f;
#pragma unroll
    for (int j = 0; j < 8; ++j) { float d = xv[j] - mean; v2 += d * d; }
    for (int o = 16; o; o >>= 1) v2 += __shfl_xor(v2, o, 32);
    float inv = rsqrtf(v2 * (1.f / 256.f) + 1e-5f);
#pragma unroll
    for (int j = 0; j < 8; ++j) {
      int n = lane + 32 * j;
      buf[m * 256 + n] = (xv[j] - mean) * inv * g[j] + b[j];
    }
  }
}

// ---------------------------------------------------------------------------
// Kernel 1: pack weights (f32 -> bf16, transposed to (N,K) row-major) + tau.
// tau is stored as 1 / (softplus(tau) + 1) so the scan needs no division.
// ---------------------------------------------------------------------------
struct PackArgs {
  const float* Win0[5]; const float* Wg0[5]; const float* Wrec0[5];
  const float* Win1[5]; const float* Wg1[5]; const float* Wrec1[5];
  const float* tauS0[5]; const float* tauS1[5];
  unsigned short* fix0[5]; unsigned short* rec0[5];
  unsigned short* fix1[5]; unsigned short* rec1[5];
  float* tau0[5]; float* tau1[5];
};

__global__ void pack_kernel(PackArgs P) {
  int tf = blockIdx.y;
  int which = blockIdx.z;
  int idx = blockIdx.x * 256 + threadIdx.x;
  if (which == 0) {              // fix0: N=512, K=32 (zero-padded K>=16)
    if (idx >= 512 * 32) return;
    int n = idx >> 5, k = idx & 31;
    float v = 0.f;
    if (k < 16) v = (n < 256) ? P.Win0[tf][k * 256 + n]
                              : P.Wg0[tf][k * 256 + (n - 256)];
    P.fix0[tf][idx] = f2bf_bits(v);
  } else if (which == 1) {       // rec0: [Wg0_h | Wrec0], N=512, K=256
    int n = idx >> 8, k = idx & 255;
    float v = (n < 256) ? P.Wg0[tf][(16 + k) * 256 + n]
                        : P.Wrec0[tf][k * 256 + (n - 256)];
    P.rec0[tf][idx] = f2bf_bits(v);
  } else if (which == 2) {       // fix1: [Win1 | Wg1_u]
    int n = idx >> 8, k = idx & 255;
    float v = (n < 256) ? P.Win1[tf][k * 256 + n]
                        : P.Wg1[tf][k * 256 + (n - 256)];
    P.fix1[tf][idx] = f2bf_bits(v);
  } else if (which == 3) {       // rec1: [Wg1_h | Wrec1]
    int n = idx >> 8, k = idx & 255;
    float v = (n < 256) ? P.Wg1[tf][(256 + k) * 256 + n]
                        : P.Wrec1[tf][k * 256 + (n - 256)];
    P.rec1[tf][idx] = f2bf_bits(v);
  } else {                       // tau: 1 / (softplus + 1)
    if (idx >= 512) return;
    const float* src = (idx < 256) ? P.tauS0[tf] : P.tauS1[tf];
    float x = src[idx & 255];
    float sp = (x > 20.f) ? x : log1pf(__expf(x));
    float* dst = (idx < 256) ? P.tau0[tf] : P.tau1[tf];
    dst[idx & 255] = 1.f / (sp + 1.f);
  }
}

// ---------------------------------------------------------------------------
// Kernel 2: LTC encoder scan with WMMA + online-softmax attention pooling.
// Grid: 5 timeframes x 16 batch blocks (16 rows each), 256 threads (8 waves).
// ---------------------------------------------------------------------------
struct EncArgs {
  const float* x[5];
  const float* bg0[5]; const float* bg1[5];
  const float* lng0[5]; const float* lnb0[5];
  const float* lng1[5]; const float* lnb1[5];
  const float* poolw[5]; const float* poolb[5];
  const unsigned short* fix0[5]; const unsigned short* rec0[5];
  const unsigned short* fix1[5]; const unsigned short* rec1[5];
  const float* tau0[5]; const float* tau1[5];
  float* enc;      // (5, 256, 256) f32
  int S[5];
};

__global__ __launch_bounds__(256) void enc_kernel(EncArgs A) {
  __shared__ unsigned short bufA[16 * 256];    // bf16 A-matrix double buffer
  __shared__ unsigned short bufB[16 * 256];
  __shared__ unsigned short u1buf[16 * 256];   // bf16 u1 (cell1 fixed-GEMM A)
  __shared__ unsigned short ubuf[16 * 32];     // bf16 input (K padded to 32)
  __shared__ float h0s[16 * 256];
  __shared__ float h1s[16 * 256];
  __shared__ float mrun[16], denom[16], scal[16], wgt[16];

  int tf = blockIdx.x >> 4;
  int mb = blockIdx.x & 15;
  int tid = threadIdx.x;
  int w = tid >> 5, lane = tid & 31;
  int m0 = mb * 16;

  for (int i = tid; i < 16 * 256; i += 256) { h0s[i] = 0.f; h1s[i] = 0.f; }
  if (tid < 16) { mrun[tid] = -1e30f; denom[tid] = 0.f; }

  // ---- hoist all loop-invariant per-lane parameters into registers ----
  int n0 = (w << 5) + (lane & 15);
  int n1 = n0 + 16;
  int mbase = (lane >> 4) << 3;
  float bg0v[2] = {A.bg0[tf][n0], A.bg0[tf][n0 + 16]};
  float bg1v[2] = {A.bg1[tf][n0], A.bg1[tf][n0 + 16]};
  float ti0[2]  = {A.tau0[tf][n0], A.tau0[tf][n0 + 16]};   // reciprocal tau
  float ti1[2]  = {A.tau1[tf][n0], A.tau1[tf][n0 + 16]};
  float g0v[8], b0v[8], g1v[8], b1v[8], pwv[8];
#pragma unroll
  for (int j = 0; j < 8; ++j) {
    int n = lane + 32 * j;
    g0v[j] = A.lng0[tf][n]; b0v[j] = A.lnb0[tf][n];
    g1v[j] = A.lng1[tf][n]; b1v[j] = A.lnb1[tf][n];
    pwv[j] = A.poolw[tf][n];
  }
  float pbv = A.poolb[tf][0];
  float acE[2][8];
#pragma unroll
  for (int t = 0; t < 2; ++t)
#pragma unroll
    for (int r = 0; r < 8; ++r) acE[t][r] = 0.f;
  __syncthreads();

  int S = A.S[tf];
  const float* x = A.x[tf];
  float fin0[2][8], fg0[2][8], fin1[2][8], fg1[2][8];
  float hb0[2][8], hb1[2][8];

  for (int s = 0; s < S; ++s) {
    // ---- cell 0 inputs: u -> ubuf, hh0 (= h0 state) -> bufA ----
    {
      int m = tid >> 4, f = tid & 15;
      float uv = x[((size_t)(m0 + m) * S + s) * 16 + f];
      ubuf[m * 32 + f] = f2bf_bits(uv);
      ubuf[m * 32 + 16 + f] = 0;
    }
#pragma unroll
    for (int t = 0; t < 2; ++t) {
      int n = t ? n1 : n0;
#pragma unroll
      for (int r = 0; r < 8; ++r) hb0[t][r] = h0s[(mbase + r) * 256 + n];
    }
    write_hh(bufA, hb0, n0, n1, mbase);
    __syncthreads();

    fixed_gemm(ubuf, 32, A.fix0[tf], 32, bg0v, w, lane, fin0, fg0);
    rk4_cell(bufA, bufB, h0s, A.rec0[tf], ti0, fin0, fg0, hb0, w, lane);
    layernorm16(h0s, g0v, b0v, w, lane);
    __syncthreads();

    // ---- cell 1 inputs: u1 = LN(h0) -> u1buf, hh0 (= h1 state) -> bufA ----
    for (int i = tid; i < 16 * 256; i += 256) u1buf[i] = f2bf_bits(h0s[i]);
#pragma unroll
    for (int t = 0; t < 2; ++t) {
      int n = t ? n1 : n0;
#pragma unroll
      for (int r = 0; r < 8; ++r) hb1[t][r] = h1s[(mbase + r) * 256 + n];
    }
    write_hh(bufA, hb1, n0, n1, mbase);
    __syncthreads();

    fixed_gemm(u1buf, 256, A.fix1[tf], 256, bg1v, w, lane, fin1, fg1);
    rk4_cell(bufA, bufB, h1s, A.rec1[tf], ti1, fin1, fg1, hb1, w, lane);
    layernorm16(h1s, g1v, b1v, w, lane);
    __syncthreads();

    // ---- online softmax attention pooling over time ----
#pragma unroll
    for (int rr = 0; rr < 2; ++rr) {
      int m = 2 * w + rr;
      float sc = 0.f;
#pragma unroll
      for (int j = 0; j < 8; ++j) { int n = lane + 32 * j; sc += h1s[m * 256 + n] * pwv[j]; }
      for (int o = 16; o; o >>= 1) sc += __shfl_xor(sc, o, 32);
      if (lane == 0) {
        sc += pbv;
        float mo = mrun[m];
        float mn = fmaxf(mo, sc);
        float e0 = __expf(mo - mn);
        float e1 = __expf(sc - mn);
        mrun[m] = mn;
        denom[m] = denom[m] * e0 + e1;
        scal[m] = e0; wgt[m] = e1;
      }
    }
    __syncthreads();
    // register accumulator update at this lane's (m, n) positions
#pragma unroll
    for (int r = 0; r < 8; ++r) {
      float sm = scal[mbase + r], wm = wgt[mbase + r];
      acE[0][r] = acE[0][r] * sm + wm * h1s[(mbase + r) * 256 + n0];
      acE[1][r] = acE[1][r] * sm + wm * h1s[(mbase + r) * 256 + n1];
    }
  }

  // ---- write enc = acc / denom ----
#pragma unroll
  for (int r = 0; r < 8; ++r) {
    float inv = 1.f / denom[mbase + r];
    size_t row = (size_t)tf * 256 + (m0 + mbase + r);
    A.enc[row * 256 + n0] = acE[0][r] * inv;
    A.enc[row * 256 + n1] = acE[1][r] * inv;
  }
}

// ---------------------------------------------------------------------------
// Kernel 3: cross-attention (5 tokens) + fusion + head. One WG per batch item.
// ---------------------------------------------------------------------------
struct FuseArgs {
  const float* enc; const float* ctx;
  const float* Wq; const float* Wk; const float* Wv; const float* Wo; const float* bo;
  const float* alng; const float* alnb;
  const float* g1w[5]; const float* g1b[5]; const float* g2w[5]; const float* g2b[5];
  const float* ctxw; const float* projw; const float* projb;
  const float* projg; const float* projbn;
  const float* hW; const float* hb;
  float* out;
};

__global__ __launch_bounds__(256) void fuse_kernel(FuseArgs P) {
  __shared__ float xb[5][256], qq[5][256], kb[5][256], vb[5][256], cx[5][256], enh[5][256];
  __shared__ float probs[4][5][5];
  __shared__ float logitsS[5], gatesS[5];
  __shared__ float zv[256], yv[256], lnsc[2];
  int b = blockIdx.x, tid = threadIdx.x, w = tid >> 5, lane = tid & 31;

  for (int t = 0; t < 5; ++t) xb[t][tid] = P.enc[((size_t)t * 256 + b) * 256 + tid];
  __syncthreads();

  {  // q, k, v projections
    int n = tid;
    for (int t = 0; t < 5; ++t) {
      float sq = 0.f, sk = 0.f, sv = 0.f;
      for (int c = 0; c < 256; ++c) {
        float xv = xb[t][c];
        sq += xv * P.Wq[c * 256 + n];
        sk += xv * P.Wk[c * 256 + n];
        sv += xv * P.Wv[c * 256 + n];
      }
      qq[t][n] = sq; kb[t][n] = sk; vb[t][n] = sv;
    }
  }
  __syncthreads();

  if (tid < 20) {  // scores + softmax per (head, query)
    int h = tid / 5, i = tid % 5;
    float sc[5]; float mx = -1e30f;
    for (int j = 0; j < 5; ++j) {
      float s = 0.f;
      for (int d = 0; d < 64; ++d) s += qq[i][h * 64 + d] * kb[j][h * 64 + d];
      s *= 0.125f;  // 1/sqrt(64)
      sc[j] = s; mx = fmaxf(mx, s);
    }
    float den = 0.f;
    for (int j = 0; j < 5; ++j) { sc[j] = __expf(sc[j] - mx); den += sc[j]; }
    for (int j = 0; j < 5; ++j) probs[h][i][j] = sc[j] / den;
  }
  __syncthreads();

  {  // context
    int n = tid, h = n >> 6;
    for (int i = 0; i < 5; ++i) {
      float s = 0.f;
      for (int j = 0; j < 5; ++j) s += probs[h][i][j] * vb[j][n];
      cx[i][n] = s;
    }
  }
  __syncthreads();

  {  // output proj + residual
    int n = tid;
    for (int i = 0; i < 5; ++i) {
      float s = P.bo[n];
      for (int c = 0; c < 256; ++c) s += cx[i][c] * P.Wo[c * 256 + n];
      enh[i][n] = xb[i][n] + s;
    }
  }
  __syncthreads();

  if (w < 5) {  // LN per token
    int i = w;
    float xv[8], s = 0.f;
    for (int j = 0; j < 8; ++j) { xv[j] = enh[i][lane + 32 * j]; s += xv[j]; }
    for (int o = 16; o; o >>= 1) s += __shfl_xor(s, o, 32);
    float mean = s * (1.f / 256.f), v2 = 0.f;
    for (int j = 0; j < 8; ++j) { float d = xv[j] - mean; v2 += d * d; }
    for (int o = 16; o; o >>= 1) v2 += __shfl_xor(v2, o, 32);
    float inv = rsqrtf(v2 * (1.f / 256.f) + 1e-5f);
    for (int j = 0; j < 8; ++j) {
      int n = lane + 32 * j;
      enh[i][n] = (xv[j] - mean) * inv * P.alng[n] + P.alnb[n];
    }
  }
  __syncthreads();

  if (w < 5) {  // fusion gate logits
    int i = w;
    float part = 0.f;
    for (int jj = lane; jj < 64; jj += 32) {
      float hsum = P.g1b[i][jj];
      for (int c = 0; c < 256; ++c) hsum += enh[i][c] * P.g1w[i][c * 64 + jj];
      part += tanhf(hsum) * P.g2w[i][jj];
    }
    for (int o = 16; o; o >>= 1) part += __shfl_xor(part, o, 32);
    if (lane == 0) logitsS[i] = part + P.g2b[i][0];
  }
  __syncthreads();

  if (tid == 0) {
    float mx = -1e30f;
    for (int i = 0; i < 5; ++i) mx = fmaxf(mx, logitsS[i]);
    float den = 0.f, e[5];
    for (int i = 0; i < 5; ++i) { e[i] = __expf(logitsS[i] - mx); den += e[i]; }
    for (int i = 0; i < 5; ++i) gatesS[i] = e[i] / den;
  }
  __syncthreads();

  {  // gated + baseline + context residual
    int n = tid;
    float g = 0.f, base = 0.f;
    for (int i = 0; i < 5; ++i) { g += enh[i][n] * gatesS[i]; base += enh[i][n]; }
    base *= 0.2f;
    float res = base;
    for (int c = 0; c < 32; ++c) res += P.ctx[(size_t)b * 32 + c] * P.ctxw[c * 256 + n];
    zv[n] = g + res;
  }
  __syncthreads();

  {  // projection
    int n = tid;
    float s = P.projb[n];
    for (int c = 0; c < 256; ++c) s += zv[c] * P.projw[c * 256 + n];
    yv[n] = s;
  }
  __syncthreads();

  if (w == 0) {  // LN stats over 256
    float xv[8], s = 0.f;
    for (int j = 0; j < 8; ++j) { xv[j] = yv[lane + 32 * j]; s += xv[j]; }
    for (int o = 16; o; o >>= 1) s += __shfl_xor(s, o, 32);
    float mean = s * (1.f / 256.f), v2 = 0.f;
    for (int j = 0; j < 8; ++j) { float d = xv[j] - mean; v2 += d * d; }
    for (int o = 16; o; o >>= 1) v2 += __shfl_xor(v2, o, 32);
    if (lane == 0) { lnsc[0] = mean; lnsc[1] = rsqrtf(v2 * (1.f / 256.f) + 1e-5f); }
  }
  __syncthreads();

  {
    int n = tid;
    zv[n] = fmaxf(0.f, (yv[n] - lnsc[0]) * lnsc[1] * P.projg[n] + P.projbn[n]);
  }
  __syncthreads();

  if (tid < 5) {  // head
    float s = P.hb[tid];
    for (int c = 0; c < 256; ++c) s += zv[c] * P.hW[c * 5 + tid];
    P.out[(size_t)b * 5 + tid] = s;
  }
}

// ---------------------------------------------------------------------------
// Launch. Input leaf order (insertion order of setup_inputs pytree):
// 0..4: x0..x4, 5: context, 6+16*tf: {tau0,Win0,Wrec0,Wg0,bg0, tau1,Win1,
// Wrec1,Wg1,bg1, lng[0],lng[1],lnb[0],lnb[1], poolw, poolb}, 86..92 attn
// (Wq,Wk,Wv,Wo,bo,lng,lnb), 93..117 fusion, 118..119 head.
// ---------------------------------------------------------------------------
extern "C" void kernel_launch(void* const* d_in, const int* in_sizes, int n_in,
                              void* d_out, int out_size, void* d_ws, size_t ws_size,
                              hipStream_t stream) {
  if (n_in < 120) return;
  auto F = [&](int i) { return (const float*)d_in[i]; };

  // workspace layout
  uint8_t* ws = (uint8_t*)d_ws;
  size_t off = 0;
  float* enc = (float*)(ws + off); off += 5ull * 256 * 256 * sizeof(float);
  unsigned short* fix0[5]; unsigned short* rec0[5];
  unsigned short* fix1[5]; unsigned short* rec1[5];
  float* tau0[5]; float* tau1[5];
  for (int tf = 0; tf < 5; ++tf) {
    fix0[tf] = (unsigned short*)(ws + off); off += 512ull * 32 * 2;
    rec0[tf] = (unsigned short*)(ws + off); off += 512ull * 256 * 2;
    fix1[tf] = (unsigned short*)(ws + off); off += 512ull * 256 * 2;
    rec1[tf] = (unsigned short*)(ws + off); off += 512ull * 256 * 2;
    tau0[tf] = (float*)(ws + off); off += 256 * sizeof(float);
    tau1[tf] = (float*)(ws + off); off += 256 * sizeof(float);
  }
  if (off > ws_size) return;

  static const int SEQS[5] = {256, 128, 64, 48, 32};

  PackArgs P{};
  EncArgs  E{};
  for (int tf = 0; tf < 5; ++tf) {
    int base = 6 + 16 * tf;
    P.tauS0[tf] = F(base + 0); P.Win0[tf] = F(base + 1);
    P.Wrec0[tf] = F(base + 2); P.Wg0[tf]  = F(base + 3);
    P.tauS1[tf] = F(base + 5); P.Win1[tf] = F(base + 6);
    P.Wrec1[tf] = F(base + 7); P.Wg1[tf]  = F(base + 8);
    P.fix0[tf] = fix0[tf]; P.rec0[tf] = rec0[tf];
    P.fix1[tf] = fix1[tf]; P.rec1[tf] = rec1[tf];
    P.tau0[tf] = tau0[tf]; P.tau1[tf] = tau1[tf];

    E.x[tf]    = F(tf);
    E.bg0[tf]  = F(base + 4);  E.bg1[tf]  = F(base + 9);
    E.lng0[tf] = F(base + 10); E.lng1[tf] = F(base + 11);
    E.lnb0[tf] = F(base + 12); E.lnb1[tf] = F(base + 13);
    E.poolw[tf] = F(base + 14); E.poolb[tf] = F(base + 15);
    E.fix0[tf] = fix0[tf]; E.rec0[tf] = rec0[tf];
    E.fix1[tf] = fix1[tf]; E.rec1[tf] = rec1[tf];
    E.tau0[tf] = tau0[tf]; E.tau1[tf] = tau1[tf];
    E.S[tf] = SEQS[tf];
  }
  E.enc = enc;

  FuseArgs G{};
  G.enc = enc; G.ctx = F(5);
  G.Wq = F(86); G.Wk = F(87); G.Wv = F(88); G.Wo = F(89); G.bo = F(90);
  G.alng = F(91); G.alnb = F(92);
  for (int i = 0; i < 5; ++i) {
    G.g1w[i] = F(93 + i); G.g1b[i] = F(98 + i);
    G.g2w[i] = F(103 + i); G.g2b[i] = F(108 + i);
  }
  G.ctxw = F(113); G.projw = F(114); G.projb = F(115);
  G.projg = F(116); G.projbn = F(117);
  G.hW = F(118); G.hb = F(119);
  G.out = (float*)d_out;

  pack_kernel<<<dim3(512, 5, 5), 256, 0, stream>>>(P);
  enc_kernel<<<dim3(80), 256, 0, stream>>>(E);
  fuse_kernel<<<dim3(256), 256, 0, stream>>>(G);
  (void)in_sizes; (void)out_size;
}
